// GRU_13907104105002
// MI455X (gfx1250) — matrix operand order repaired and checked by reading, not compiled, not run on hardware
//
#include <hip/hip_runtime.h>
#include <cstdint>
#include <cstddef>

typedef _Float16 v16h __attribute__((ext_vector_type(16)));
typedef __fp16   fp16x2 __attribute__((ext_vector_type(2)));   // cvt_pkrtz return type
typedef float    v8f  __attribute__((ext_vector_type(8)));
typedef float    v2f  __attribute__((ext_vector_type(2)));

constexpr int T_ = 200;
constexpr int B_ = 4096;
constexpr int I_ = 3;
constexpr int H_ = 32;
constexpr int TILES_PER_DIR = B_ / 16;          // 256
constexpr int NBLOCKS = 2 * TILES_PER_DIR;      // 512 single-wave workgroups

#if __has_builtin(__builtin_amdgcn_wmma_f32_16x16x4_f32)
#define USE_F32X4 1
#else
#define USE_F32X4 0
#endif

__device__ __forceinline__ float fast_exp(float x) {
#if __has_builtin(__builtin_amdgcn_exp2f)
  return __builtin_amdgcn_exp2f(x * 1.44269504088896340736f);
#else
  return __expf(x);
#endif
}
__device__ __forceinline__ float fast_rcp(float x) {
#if __has_builtin(__builtin_amdgcn_rcpf)
  return __builtin_amdgcn_rcpf(x);
#else
  return 1.0f / x;
#endif
}
__device__ __forceinline__ float sigmoid_(float x) {
  return fast_rcp(1.0f + fast_exp(-x));
}
__device__ __forceinline__ float tanh_(float x) {
#if __has_builtin(__builtin_amdgcn_tanhf)
  return __builtin_amdgcn_tanhf(x);   // v_tanh_f32 (TRANS, co-executes with VALU)
#else
  float e = fast_exp(-2.0f * x);
  return 1.0f - 2.0f * fast_rcp(1.0f + e);
#endif
}

__device__ __forceinline__ v8f wmma_f16(v16h a, v16h b, v8f c) {
  return __builtin_amdgcn_wmma_f32_16x16x32_f16(false, a, false, b, (short)0, c, false, false);
}

// out[t,b] starts as fc_b; both direction waves atomically add their FC partials.
__global__ void init_out_kernel(float* __restrict__ out, const float* __restrict__ fc_b, int n) {
  int i = blockIdx.x * blockDim.x + threadIdx.x;
  if (i < n) out[i] = fc_b[0];
}

// One wave (= one single-wave workgroup) = one (direction, 16-batch tile).
// Entire recurrence in registers:
//   hg^T[96x16] = W_hh[96x32] (A, f16, resident) x h^T[32x16] (B, rebuilt per step)
// Input-gate term seeded via f32 16x16x4 WMMA with K = {x0,x1,x2,1.0}, biases folded
// into the A matrix's K=3 column. FC output fused via per-lane partials + SWAPX16.
__global__ void __launch_bounds__(32, 1) gru_bidir_kernel(
    const float* __restrict__ x,
    const float* __restrict__ wih_f, const float* __restrict__ whh_f,
    const float* __restrict__ bih_f, const float* __restrict__ bhh_f,
    const float* __restrict__ wih_b, const float* __restrict__ whh_b,
    const float* __restrict__ bih_b, const float* __restrict__ bhh_b,
    const float* __restrict__ fc_w,
    float* __restrict__ out)
{
  const int lane = threadIdx.x & 31;
  const int gid  = blockIdx.x;
  const int dir  = gid >> 8;                      // 0 = forward, 1 = backward
  const int tile = gid & (TILES_PER_DIR - 1);
  const int l    = lane & 15;
  const int hf   = lane >> 4;                     // which half-wave
  const int b    = tile * 16 + l;                 // batch column served by this lane

  const float* wih = dir ? wih_b : wih_f;
  const float* whh = dir ? whh_b : whh_f;
  const float* bih = dir ? bih_b : bih_f;
  const float* bhh = dir ? bhh_b : bhh_f;

  // Shared index map: A-operand K index == C-layout gate offset == per-lane j map.
  // jm[e] = e + 8*(e>=8) + 8*hf  (covers {0..7,16..23} / {8..15,24..31})
  int jm[16];
  #pragma unroll
  for (int e = 0; e < 16; ++e) jm[e] = e + ((e >= 8) ? 8 : 0) + 8 * hf;

  // Resident A tiles of W_hh (f16 16x32 A-layout); tile g = gate rows 16g..16g+15.
  v16h Ahh[6];
  #pragma unroll
  for (int g = 0; g < 6; ++g) {
    const int gate = 16 * g + l;
    #pragma unroll
    for (int e = 0; e < 16; ++e)
      Ahh[g][e] = (_Float16)whh[gate * H_ + jm[e]];
  }

#if USE_F32X4
  // A tiles for augmented W_ih (f32 16x4 A-layout): K0,K1 on lanes 0-15; K2,K3(bias) on 16-31.
  v2f Aih[6];
  #pragma unroll
  for (int g = 0; g < 6; ++g) {
    const int gate = 16 * g + l;
    const float bc = bih[gate] + ((g < 4) ? bhh[gate] : 0.0f);  // fold b_hh for r,z only
    Aih[g][0] = hf ? wih[gate * I_ + 2] : wih[gate * I_ + 0];
    Aih[g][1] = hf ? bc                 : wih[gate * I_ + 1];
  }
#else
  // Fallback: pad K=4 into a f16 16x32 A tile (lanes 16-31 all zero).
  v16h Aih16[6];
  #pragma unroll
  for (int g = 0; g < 6; ++g) {
    const int gate = 16 * g + l;
    const float bc = bih[gate] + ((g < 4) ? bhh[gate] : 0.0f);
    #pragma unroll
    for (int e = 0; e < 16; ++e) Aih16[g][e] = (_Float16)0.0f;
    Aih16[g][0] = (_Float16)(hf ? 0.0f : wih[gate * I_ + 0]);
    Aih16[g][1] = (_Float16)(hf ? 0.0f : wih[gate * I_ + 1]);
    Aih16[g][2] = (_Float16)(hf ? 0.0f : wih[gate * I_ + 2]);
    Aih16[g][3] = (_Float16)(hf ? 0.0f : bc);
  }
#endif

  // Per-lane constants: n-gate hidden bias and FC weights at this lane's j positions.
  float bhn[16], fcl[16];
  #pragma unroll
  for (int e = 0; e < 16; ++e) {
    bhn[e] = bhh[2 * H_ + jm[e]];
    fcl[e] = fc_w[dir * H_ + jm[e]];
  }

  float h[16];
  #pragma unroll
  for (int e = 0; e < 16; ++e) h[e] = 0.0f;

  const v8f vzero = 0.0f;

  // Software-pipelined x load for step 0.
  const int t0 = dir ? (T_ - 1) : 0;
  const float* xp0 = x + ((size_t)t0 * B_ + b) * I_;
  float x0 = xp0[0], x1 = xp0[1], x2 = xp0[2];

  #pragma unroll 1
  for (int s = 0; s < T_; ++s) {
    const int t = dir ? (T_ - 1 - s) : s;
    const float cx0 = x0, cx1 = x1, cx2 = x2;

    // Prefetch next step's x (L2-resident; overlaps with WMMA + gate math below).
    if (s + 1 < T_) {
      const int tn = dir ? (T_ - 2 - s) : (s + 1);
      const float* xpn = x + ((size_t)tn * B_ + b) * I_;
      x0 = xpn[0]; x1 = xpn[1]; x2 = xpn[2];
    }

    v8f acc[6];
#if USE_F32X4
    v2f Bx;
    Bx[0] = hf ? cx2  : cx0;   // K2 / K0
    Bx[1] = hf ? 1.0f : cx1;   // K3(bias col) / K1
    #pragma unroll
    for (int g = 0; g < 6; ++g)
      acc[g] = __builtin_amdgcn_wmma_f32_16x16x4_f32(
          false, Aih[g], false, Bx, (short)0, vzero, false, false);
#else
    v16h Bx16;
    #pragma unroll
    for (int e = 0; e < 16; ++e) Bx16[e] = (_Float16)0.0f;
    Bx16[0] = (_Float16)(hf ? 0.0f : cx0);
    Bx16[1] = (_Float16)(hf ? 0.0f : cx1);
    Bx16[2] = (_Float16)(hf ? 0.0f : cx2);
    Bx16[3] = (_Float16)(hf ? 0.0f : 1.0f);
    #pragma unroll
    for (int g = 0; g < 6; ++g)
      acc[g] = wmma_f16(Aih16[g], Bx16, vzero);
#endif

    // Build B operand h^T[32x16] (f16 B-layout: lanes 0-15 K={0..7,16..23}, lanes 16-31 K={8..15,24..31})
    // from the lane-local C-layout h via f16 pack alone, no cross-lane exchange (no LDS memory traffic).
    uint32_t pk[8], po[8];
    #pragma unroll
    for (int k = 0; k < 8; ++k) {
      union { fp16x2 h2; uint32_t u; } cv;
      cv.h2 = __builtin_amdgcn_cvt_pkrtz(h[2 * k], h[2 * k + 1]);
      pk[k] = cv.u;
    }
    #pragma unroll
    for (int k = 0; k < 8; ++k)
      po[k] = (uint32_t)__builtin_amdgcn_ds_swizzle((int)pk[k], 0x401F); // SWAPX16
    union { v16h h16; uint32_t u[8]; } hb;
    #pragma unroll
    for (int v = 0; v < 4; ++v) {
      hb.u[v]     = pk[v];
      hb.u[v + 4] = pk[v + 4];
    }

    // Recurrent matmul: r,z tiles accumulate onto xg; n tiles kept separate (b_hh_n inside r*hn).
    #pragma unroll
    for (int g = 0; g < 4; ++g)
      acc[g] = wmma_f16(Ahh[g], hb.h16, acc[g]);
    v8f hn0 = wmma_f16(Ahh[4], hb.h16, vzero);
    v8f hn1 = wmma_f16(Ahh[5], hb.h16, vzero);

    // Lane-local GRU gate math + fused FC partial.
    float p = 0.0f;
    #pragma unroll
    for (int e = 0; e < 16; ++e) {
      const float xr = (e < 8) ? acc[0][e] : acc[1][e - 8];   // xr + hr (+biases)
      const float xz = (e < 8) ? acc[2][e] : acc[3][e - 8];   // xz + hz (+biases)
      const float xn = (e < 8) ? acc[4][e] : acc[5][e - 8];   // xn + b_ih_n
      const float hn = ((e < 8) ? hn0[e] : hn1[e - 8]) + bhn[e];
      const float r  = sigmoid_(xr);
      const float z  = sigmoid_(xz);
      const float nn = tanh_(xn + r * hn);
      h[e] = nn + z * (h[e] - nn);                            // (1-z)*n + z*h
      p    = fmaf(h[e], fcl[e], p);
    }

    // Combine the two half-wave FC partials and accumulate into out.
    union { float f; int i; } pu, pv;
    pu.f = p;
    pv.i = __builtin_amdgcn_ds_swizzle(pu.i, 0x401F);
    if (hf == 0)
      atomicAdd(&out[(size_t)t * B_ + b], p + pv.f);
  }
}

extern "C" void kernel_launch(void* const* d_in, const int* in_sizes, int n_in,
                              void* d_out, int out_size, void* d_ws, size_t ws_size,
                              hipStream_t stream) {
  (void)in_sizes; (void)n_in; (void)d_ws; (void)ws_size;

  const float* x     = (const float*)d_in[0];
  const float* wih_f = (const float*)d_in[1];
  const float* whh_f = (const float*)d_in[2];
  const float* bih_f = (const float*)d_in[3];
  const float* bhh_f = (const float*)d_in[4];
  const float* wih_b = (const float*)d_in[5];
  const float* whh_b = (const float*)d_in[6];
  const float* bih_b = (const float*)d_in[7];
  const float* bhh_b = (const float*)d_in[8];
  const float* fc_w  = (const float*)d_in[9];
  const float* fc_b  = (const float*)d_in[10];
  float* out = (float*)d_out;

  const int n = T_ * B_;  // C == 1
  init_out_kernel<<<(n + 255) / 256, 256, 0, stream>>>(out, fc_b, n);
  gru_bidir_kernel<<<NBLOCKS, 32, 0, stream>>>(
      x, wih_f, whh_f, bih_f, bhh_f, wih_b, whh_b, bih_b, bhh_b, fc_w, out);
}
